// T5Attention_6519760355855
// MI455X (gfx1250) — matrix operand order repaired and checked by reading, not compiled, not run on hardware
//
#include <hip/hip_runtime.h>

// ---------------------------------------------------------------------------
// T5 attention forward, MI455X (gfx1250), wave32 + bf16 WMMA (f32 accumulate)
//   stage 1: Q/K/V = hs @ w{q,k,v}          (f32 in -> bf16 out, WMMA bf16)
//   stage 2: flash attention per (b,h,128q) (bf16 WMMA, fp32 online softmax,
//            K-tile staged by the Tensor Data Mover when available)
//   stage 3: out = ctx @ wo                 (bf16 in -> f32 out, WMMA bf16)
// Workspace: 4 x (4096*1024) bf16 buffers = 32 MiB.
// ---------------------------------------------------------------------------

typedef __attribute__((ext_vector_type(16))) __bf16        v16bf;
typedef __attribute__((ext_vector_type(8)))  float         v8f;
typedef __attribute__((ext_vector_type(4)))  unsigned int  v4u;
typedef __attribute__((ext_vector_type(4)))  unsigned int  u32x4;
typedef __attribute__((ext_vector_type(8)))  int           i32x8;
typedef __attribute__((ext_vector_type(4)))  int           i32x4;

#define DM  1024
#define NH  16
#define DKV 64
#define BB  2
#define SS  2048

// Tensor Data Mover availability (ROCm7.2 clang-22: 5-arg; clang-23: 6-arg)
#if defined(__has_builtin)
#  if __has_builtin(__builtin_amdgcn_tensor_load_to_lds) && \
      __has_builtin(__builtin_amdgcn_s_wait_tensorcnt)
#    define HAVE_TDM 1
#  endif
#endif

__device__ __forceinline__ unsigned short f2bf(float f) {
    __bf16 b = (__bf16)f;                       // RNE convert
    return __builtin_bit_cast(unsigned short, b);
}

// Load a v16bf fragment as two 16-byte chunks (ds_load_b128 / global_load_b128)
__device__ __forceinline__ v16bf load_frag(const unsigned short* p0,
                                           const unsigned short* p1) {
    union { v16bf v; v4u q[2]; } f;
    f.q[0] = *(const v4u*)p0;
    f.q[1] = *(const v4u*)p1;
    return f.v;
}

// ---------------------------------------------------------------------------
// Generic tiled GEMM: C[M,N] = A[M,K] * W[K,N]
//   block tile 128x128, k-step 32, 8 waves, wave tile 64x32 (4x2 WMMA tiles)
// ---------------------------------------------------------------------------
template<bool A_IS_F32, bool OUT_IS_F32>
__global__ __launch_bounds__(256)
void gemm_bf16_wmma(const void* __restrict__ Ap, const float* __restrict__ W,
                    void* __restrict__ Cp, int M, int N, int K)
{
    __shared__ unsigned short lds_a[128][40];   // [m][k], 80B rows (16B align)
    __shared__ unsigned short lds_b[128][40];   // [n][k] (transposed W tile)

    const int tid  = threadIdx.x;
    const int wid  = tid >> 5;
    const int lane = tid & 31;
    const int l15  = lane & 15;
    const int kh   = lane >> 4;                 // lane half selects K sub-range

    const int bm = blockIdx.y * 128;
    const int bn = blockIdx.x * 128;
    const int wm = (wid >> 2) * 64;
    const int wn = (wid & 3) * 32;

    const v8f vz = {0.f,0.f,0.f,0.f,0.f,0.f,0.f,0.f};
    v8f acc[4][2];
    #pragma unroll
    for (int i = 0; i < 4; ++i)
        #pragma unroll
        for (int j = 0; j < 2; ++j) acc[i][j] = vz;

    for (int k0 = 0; k0 < K; k0 += 32) {
        // A tile 128x32 -> bf16 LDS: 8-element chunks, b128 LDS stores
        #pragma unroll
        for (int i = 0; i < 2; ++i) {
            int c = tid + i * 256;              // 512 chunks of 8
            int r = c >> 2, cc = (c & 3) * 8;
            if constexpr (A_IS_F32) {
                const float* ap = (const float*)Ap + (size_t)(bm + r) * K + k0 + cc;
                union { unsigned short h[8]; v4u q; } u;
                #pragma unroll
                for (int j = 0; j < 8; ++j) u.h[j] = f2bf(ap[j]);
                *(v4u*)&lds_a[r][cc] = u.q;
            } else {
                *(v4u*)&lds_a[r][cc] = *(const v4u*)
                    ((const unsigned short*)Ap + (size_t)(bm + r) * K + k0 + cc);
            }
        }
        // W tile 32x128 -> transposed [n][k]; 8 k-elems per chunk, b128 store
        #pragma unroll
        for (int i = 0; i < 2; ++i) {
            int c = tid + i * 256;
            int n = c & 127, kc = (c >> 7) * 8;
            union { unsigned short h[8]; v4u q; } u;
            #pragma unroll
            for (int j = 0; j < 8; ++j)
                u.h[j] = f2bf(W[(size_t)(k0 + kc + j) * N + bn + n]);
            *(v4u*)&lds_b[n][kc] = u.q;
        }
        __syncthreads();

        v16bf af[4], bfr[2];
        #pragma unroll
        for (int mt = 0; mt < 4; ++mt) {
            const unsigned short* row = lds_a[wm + mt * 16 + l15];
            // A 16x32 bf16 layout: half0 K0-7 & 16-23, half1 K8-15 & 24-31
            af[mt] = load_frag(row + kh * 8, row + 16 + kh * 8);
        }
        #pragma unroll
        for (int nt = 0; nt < 2; ++nt) {
            const unsigned short* row = lds_b[wn + nt * 16 + l15];
            // B 32x16 layout: half0 K0-7 & 16-23, half1 K8-15 & 24-31
            bfr[nt] = load_frag(row + kh * 8, row + 16 + kh * 8);
        }
        #pragma unroll
        for (int mt = 0; mt < 4; ++mt)
            #pragma unroll
            for (int nt = 0; nt < 2; ++nt)
                acc[mt][nt] = __builtin_amdgcn_wmma_f32_16x16x32_bf16(
                    false, af[mt], false, bfr[nt], (short)0, acc[mt][nt],
                    false, false);
        __syncthreads();
    }

    // C layout: VGPR r -> row r (lanes 0-15) / row r+8 (lanes 16-31)
    #pragma unroll
    for (int mt = 0; mt < 4; ++mt)
        #pragma unroll
        for (int nt = 0; nt < 2; ++nt)
            #pragma unroll
            for (int r = 0; r < 8; ++r) {
                int row = bm + wm + mt * 16 + r + kh * 8;
                int col = bn + wn + nt * 16 + l15;
                float v = acc[mt][nt][r];
                if constexpr (OUT_IS_F32)
                    ((float*)Cp)[(size_t)row * N + col] = v;
                else
                    ((unsigned short*)Cp)[(size_t)row * N + col] = f2bf(v);
            }
}

// ---------------------------------------------------------------------------
// Flash attention: one block per (b, h, 128 q-rows); 8 waves x 16 q-rows.
// Q,K,V,Ctx are bf16 [B*S, H*DKV] row-major; head h occupies cols [64h,64h+64)
// ---------------------------------------------------------------------------
__global__ __launch_bounds__(256)
void attn_flash_wmma(const unsigned short* __restrict__ Q,
                     const unsigned short* __restrict__ Kb,
                     const unsigned short* __restrict__ V,
                     unsigned short* __restrict__ Ctx)
{
    __shared__ unsigned short lds_k [128][72];    // [key][dim], 64 data + 8 pad
    __shared__ unsigned short lds_vt[ 64][136];   // [dim][key]  (V transposed)
    __shared__ unsigned short lds_p [8][16][136]; // per-wave P tile [m][key]

    const int tid  = threadIdx.x;
    const int wid  = tid >> 5;
    const int lane = tid & 31;
    const int l15  = lane & 15;
    const int kh   = lane >> 4;
    const int qb   = blockIdx.x;     // q block (128 rows)
    const int h    = blockIdx.y;
    const int b    = blockIdx.z;

    const size_t head_off = (size_t)h * DKV;

    // Q fragments for this wave's 16 rows (A-matrix layout, two 32-dim chunks)
    v16bf qf[2];
    {
        const unsigned short* qp =
            Q + ((size_t)b * SS + qb * 128 + wid * 16 + l15) * DM + head_off;
        qf[0] = load_frag(qp + kh * 8,       qp + 16 + kh * 8);
        qf[1] = load_frag(qp + 32 + kh * 8,  qp + 48 + kh * 8);
    }

    float m[8], l[8];
    #pragma unroll
    for (int r = 0; r < 8; ++r) { m[r] = -__builtin_inff(); l[r] = 0.f; }
    const v8f vz = {0.f,0.f,0.f,0.f,0.f,0.f,0.f,0.f};
    v8f ctxa[4] = {vz, vz, vz, vz};

    for (int kb = 0; kb < SS / 128; ++kb) {
        __syncthreads();

        // ---- K block -> LDS [key][64 data + 8 pad halves] ----
#if defined(HAVE_TDM)
        if (wid == 0) {
            // Tensor DMA: 2D tile 64x128 of 2B elems out of tensor 1024x4096,
            // LDS pad: +4 DWORDs after every 32 DWORDs (row stride 72 halves).
            unsigned long long ga = (unsigned long long)(const void*)
                (Kb + ((size_t)b * SS + kb * 128) * DM + head_off);
            unsigned lds_addr = (unsigned)(unsigned long long)&lds_k[0][0];
            u32x4 g0;
            g0[0] = 1u;                                   // count=1 (user D#)
            g0[1] = lds_addr;                             // lds_addr
            g0[2] = (unsigned)(ga & 0xFFFFFFFFull);       // global_addr[31:0]
            g0[3] = (unsigned)((ga >> 32) & 0x01FFFFFFull)// global_addr[56:32]
                    | 0x80000000u;                        // type=2 ("image")
            i32x8 g1;
            g1[0] = (int)0x07110000;   // data_size=2B, pad_en, int=32dw, amt=4dw
            g1[1] = (int)(((unsigned)DM & 0xFFFFu) << 16);        // tensor_dim0
            g1[2] = (int)((((unsigned)(BB * SS)) & 0xFFFFu) << 16 // tensor_dim1
                          | ((unsigned)DM >> 16));
            g1[3] = (int)(((unsigned)DKV) << 16);                 // tile_dim0=64
            g1[4] = (int)128;                                     // tile_dim1
            g1[5] = (int)DM;                                      // dim0_stride
            g1[6] = 0; g1[7] = 0;
            i32x4 g2 = {0, 0, 0, 0};
            i32x4 g3 = {0, 0, 0, 0};
#if __clang_major__ >= 23
            i32x8 g4 = {0, 0, 0, 0, 0, 0, 0, 0};
            __builtin_amdgcn_tensor_load_to_lds(g0, g1, g2, g3, g4, 0);
#else
            __builtin_amdgcn_tensor_load_to_lds(g0, g1, g2, g3, 0);
#endif
        }
#else
        #pragma unroll
        for (int i = 0; i < 4; ++i) {
            int e = tid + i * 256;
            int key = e >> 3, dc = e & 7;
            const v4u* src = (const v4u*)(Kb +
                ((size_t)b * SS + kb * 128 + key) * DM + head_off + dc * 8);
            *(v4u*)&lds_k[key][dc * 8] = *src;
        }
#endif
        // ---- V block -> LDS transposed [dim][key] ----
        #pragma unroll
        for (int i = 0; i < 32; ++i) {
            int e = tid + i * 256;
            int key = e >> 6, d = e & 63;
            lds_vt[d][key] =
                V[((size_t)b * SS + kb * 128 + key) * DM + head_off + d];
        }
#if defined(HAVE_TDM)
        if (wid == 0) __builtin_amdgcn_s_wait_tensorcnt(0);
#endif
        __syncthreads();

        // scores S = Q @ K^T : 16 q-rows x 128 keys (8 tiles, 2 k-steps each)
        v8f s[8];
        #pragma unroll
        for (int nt = 0; nt < 8; ++nt) {
            const unsigned short* krow = lds_k[nt * 16 + l15];
            v16bf b0 = load_frag(krow + kh * 8,      krow + 16 + kh * 8);
            v16bf b1 = load_frag(krow + 32 + kh * 8, krow + 48 + kh * 8);
            v8f a = vz;
            a = __builtin_amdgcn_wmma_f32_16x16x32_bf16(
                    false, qf[0], false, b0, (short)0, a, false, false);
            a = __builtin_amdgcn_wmma_f32_16x16x32_bf16(
                    false, qf[1], false, b1, (short)0, a, false, false);
            s[nt] = a;
        }

        // --- online softmax, fp32 (T5: no 1/sqrt(d) scaling) ---
        float rmax[8];
        #pragma unroll
        for (int r = 0; r < 8; ++r) {
            float v = s[0][r];
            #pragma unroll
            for (int nt = 1; nt < 8; ++nt) v = fmaxf(v, s[nt][r]);
            #pragma unroll
            for (int x = 8; x >= 1; x >>= 1)
                v = fmaxf(v, __shfl_xor(v, x, 32));   // row lives in 16-lane half
            rmax[r] = v;
        }
        float sc[8];
        #pragma unroll
        for (int r = 0; r < 8; ++r) {
            float mn = fmaxf(m[r], rmax[r]);
            sc[r] = __expf(m[r] - mn);
            m[r]  = mn;
        }
        float rsum[8];
        #pragma unroll
        for (int r = 0; r < 8; ++r) rsum[r] = 0.f;
        #pragma unroll
        for (int nt = 0; nt < 8; ++nt)
            #pragma unroll
            for (int r = 0; r < 8; ++r) {
                float p = __expf(s[nt][r] - m[r]);
                s[nt][r] = p;
                rsum[r] += p;
            }
        #pragma unroll
        for (int r = 0; r < 8; ++r) {
            float v = rsum[r];
            #pragma unroll
            for (int x = 8; x >= 1; x >>= 1) v += __shfl_xor(v, x, 32);
            l[r] = l[r] * sc[r] + v;
        }
        #pragma unroll
        for (int dt = 0; dt < 4; ++dt)
            #pragma unroll
            for (int r = 0; r < 8; ++r) ctxa[dt][r] *= sc[r];

        // P (C-layout) -> per-wave LDS row-major [m][key] as bf16
        #pragma unroll
        for (int nt = 0; nt < 8; ++nt)
            #pragma unroll
            for (int r = 0; r < 8; ++r)
                lds_p[wid][r + kh * 8][nt * 16 + l15] = f2bf(s[nt][r]);
        __builtin_amdgcn_wave_barrier();  // keep DS store->load program order

        // ctx += P @ V   (4 k-chunks of 32 keys x 4 dim tiles)
        #pragma unroll
        for (int kc = 0; kc < 4; ++kc) {
            const unsigned short* prow = lds_p[wid][l15];
            v16bf pf = load_frag(prow + kc * 32 + kh * 8,
                                 prow + kc * 32 + 16 + kh * 8);
            #pragma unroll
            for (int dt = 0; dt < 4; ++dt) {
                const unsigned short* vrow = lds_vt[dt * 16 + l15];
                v16bf vf = load_frag(vrow + kc * 32 + kh * 8,
                                     vrow + kc * 32 + 16 + kh * 8);
                ctxa[dt] = __builtin_amdgcn_wmma_f32_16x16x32_bf16(
                    false, pf, false, vf, (short)0, ctxa[dt], false, false);
            }
        }
    }

    // normalize and store ctx (bf16)
    #pragma unroll
    for (int dt = 0; dt < 4; ++dt)
        #pragma unroll
        for (int r = 0; r < 8; ++r) {
            float v = ctxa[dt][r] / l[r];
            size_t row = (size_t)b * SS + qb * 128 + wid * 16 + r + kh * 8;
            size_t col = head_off + dt * 16 + l15;
            Ctx[row * DM + col] = f2bf(v);
        }
}

// ---------------------------------------------------------------------------
extern "C" void kernel_launch(void* const* d_in, const int* in_sizes, int n_in,
                              void* d_out, int out_size, void* d_ws, size_t ws_size,
                              hipStream_t stream)
{
    const float* hs = (const float*)d_in[0];   // [B,S,DM]
    const float* wq = (const float*)d_in[1];   // [DM, NH*DKV]
    const float* wk = (const float*)d_in[2];
    const float* wv = (const float*)d_in[3];
    const float* wo = (const float*)d_in[4];   // [NH*DKV, DM]
    float* out = (float*)d_out;                // [B,S,DM] fp32

    const size_t TOK = (size_t)BB * SS;        // 4096 rows
    unsigned short* qbuf = (unsigned short*)d_ws;
    unsigned short* kbuf = qbuf + TOK * DM;
    unsigned short* vbuf = kbuf + TOK * DM;
    unsigned short* cbuf = vbuf + TOK * DM;    // needs 32 MiB total

    dim3 blk(256);
    dim3 gg(DM / 128, TOK / 128);              // (8, 32)

    gemm_bf16_wmma<true,  false><<<gg, blk, 0, stream>>>(hs, wq, qbuf,
                                                         (int)TOK, DM, DM);
    gemm_bf16_wmma<true,  false><<<gg, blk, 0, stream>>>(hs, wk, kbuf,
                                                         (int)TOK, DM, DM);
    gemm_bf16_wmma<true,  false><<<gg, blk, 0, stream>>>(hs, wv, vbuf,
                                                         (int)TOK, DM, DM);

    dim3 ga(SS / 128, NH, BB);                 // (16, 16, 2)
    attn_flash_wmma<<<ga, blk, 0, stream>>>(qbuf, kbuf, vbuf, cbuf);

    gemm_bf16_wmma<false, true ><<<gg, blk, 0, stream>>>(cbuf, wo, out,
                                                         (int)TOK, DM, DM);
}
